// FeatureFusionModule_45861660787158
// MI455X (gfx1250) — compile-verified, matrix operand order checked by reading, not run on hardware
//
#include <hip/hip_runtime.h>
#include <hip/hip_bf16.h>
#include <math.h>

typedef _Float16 h16;
typedef _Float16 v8h  __attribute__((ext_vector_type(8)));
typedef _Float16 v16h __attribute__((ext_vector_type(16)));
typedef float    v8f  __attribute__((ext_vector_type(8)));
typedef int      v4i  __attribute__((ext_vector_type(4)));

#define TILE_M 128
#define TILE_N 128
#define TILE_K 32
#define LDS_PITCH 40   // 32 + 8 halves pad -> 80B row pitch, conflict-free b128 reads

// CDNA5 async global->LDS copy path (ASYNCcnt-tracked), guarded so a missing
// builtin degrades to the verified sync staging path instead of failing.
#if defined(__HIP_DEVICE_COMPILE__) && defined(__gfx1250__) && \
    __has_builtin(__builtin_amdgcn_global_load_async_to_lds_b128) && \
    __has_builtin(__builtin_amdgcn_s_wait_asynccnt)
#define FFM_ASYNC 1
#define FFM_AS1 __attribute__((address_space(1)))
#define FFM_AS3 __attribute__((address_space(3)))
#else
#define FFM_ASYNC 0
#endif

enum {
    EPI_BIAS  = 1,
    EPI_GELU  = 2,
    EPI_EXP   = 4,
    EPI_DIV   = 8,
    EPI_RES   = 16,
    EPI_OUT32 = 32,
    EPI_OUT16 = 64
};

struct GemmParams {
    const h16*   A;
    const h16*   B;      // stored (N, K) row-major unless transB
    float*       C32;
    h16*         C16;
    const float* bias;
    const float* rowAux; // EXP: subtract per-row; DIV: divide per-row
    const float* res;
    int M, N, K;
    int lda, ldb, ldc32, ldc16, ldres;
    long long strideA, strideB, strideC32, strideC16, strideAux;
    int transA, transB, flags, c32off;
    float scale;
};

__device__ __forceinline__ float ffm_gelu(float x) {
    return 0.5f * x * (1.0f + erff(x * 0.70710678118654752f));
}

// Stage one (128 x 32) tile of logical (row, k) data into LDS.
// trans==1 means source is stored (K, rows) row-major.
__device__ __forceinline__ void ffm_stage_tile(const h16* __restrict__ src,
                                               h16* __restrict__ sm,
                                               int base, int limit, int ld,
                                               int kt, int K, int trans, int tid) {
    if (!trans) {
        const bool full = (base + TILE_M) <= limit;
#if FFM_ASYNC
        if (full) {
            // Per-lane 16B DMA directly into LDS; no VGPR staging.
            #pragma unroll
            for (int p = 0; p < 2; ++p) {
                int q  = tid + (p << 8);
                int r  = q >> 2;
                int cg = (q & 3) << 3;
                const h16* gp = src + (long long)(base + r) * ld + kt + cg;
                __builtin_amdgcn_global_load_async_to_lds_b128(
                    (FFM_AS1 v4i*)gp,
                    (FFM_AS3 v4i*)(&sm[r * LDS_PITCH + cg]), 0, 0);
            }
            return;
        }
#endif
        #pragma unroll
        for (int p = 0; p < 2; ++p) {
            int q  = tid + (p << 8);
            int r  = q >> 2;
            int cg = (q & 3) << 3;
            int gr = base + r;
            v8h v = {};
            if (full || gr < limit) {
                const h16* gp = src + (long long)gr * ld + kt + cg;
                v = *(const v8h*)gp;
                if (kt + TILE_K < K)
                    __builtin_prefetch(gp + TILE_K, 0, 1);
            }
            *(v8h*)(&sm[r * LDS_PITCH + cg]) = v;
        }
    } else {
        // element (r, k) = src[(kt+k)*ld + base + r]; read 8 contiguous rows
        #pragma unroll
        for (int p = 0; p < 2; ++p) {
            int q  = tid + (p << 8);
            int c  = q & 31;          // k within tile
            int r0 = (q >> 5) << 3;   // row group 0..120
            v8h v = {};
            if (base + r0 < limit)
                v = *(const v8h*)(src + (long long)(kt + c) * ld + base + r0);
            #pragma unroll
            for (int j = 0; j < 8; ++j)
                sm[(r0 + j) * LDS_PITCH + c] = v[j];
        }
    }
}

__device__ __forceinline__ void ffm_stage_fence() {
#if FFM_ASYNC
    __builtin_amdgcn_s_wait_asynccnt(0);
#endif
    __syncthreads();
}

// A/B fragment per documented 16-bit 16x32 layout:
// lanes 0-15 hold K=0..7 (VGPR0-3) and K=16..23 (VGPR4-7); lanes 16-31: +8.
__device__ __forceinline__ v16h ffm_load_frag(const h16* __restrict__ sm,
                                              int row, int k0) {
    v8h lo = *(const v8h*)(sm + row * LDS_PITCH + k0);
    v8h hi = *(const v8h*)(sm + row * LDS_PITCH + k0 + 16);
    v16h f;
    #pragma unroll
    for (int j = 0; j < 8; ++j) { f[j] = lo[j]; f[j + 8] = hi[j]; }
    return f;
}

__global__ __launch_bounds__(256)
void ffm_gemm_wmma(GemmParams p) {
    __shared__ h16 smA[2][TILE_M * LDS_PITCH];
    __shared__ h16 smB[2][TILE_N * LDS_PITCH];

    const int tid  = threadIdx.x;
    const int wave = tid >> 5;
    const int lane = tid & 31;
    const int l16  = lane & 15;
    const int half = lane >> 4;
    const int wrow = wave >> 1;   // 0..3  -> 32 rows each
    const int wcol = wave & 1;    // 0..1  -> 64 cols each
    const int i0 = blockIdx.y * TILE_M;
    const int j0 = blockIdx.x * TILE_N;
    const int b  = blockIdx.z;

    const h16* A = p.A + (long long)b * p.strideA;
    const h16* B = p.B + (long long)b * p.strideB;
    float* C32 = p.C32 ? p.C32 + (long long)b * p.strideC32 : nullptr;
    h16*   C16 = p.C16 ? p.C16 + (long long)b * p.strideC16 : nullptr;
    const float* rowAux = p.rowAux ? p.rowAux + (long long)b * p.strideAux : nullptr;

    v8f acc[2][4];
    #pragma unroll
    for (int ai = 0; ai < 2; ++ai)
        #pragma unroll
        for (int bj = 0; bj < 4; ++bj)
            acc[ai][bj] = (v8f){};

    const int k0sel = half ? 8 : 0;

    // Double-buffered pipeline: stage K-slab kt+32 while WMMAs run on slab kt.
    int cur = 0;
    ffm_stage_tile(A, smA[0], i0, p.M, p.lda, 0, p.K, p.transA, tid);
    ffm_stage_tile(B, smB[0], j0, p.N, p.ldb, 0, p.K, p.transB, tid);
    ffm_stage_fence();

    for (int kt = 0; kt < p.K; kt += TILE_K) {
        const int nxt = cur ^ 1;
        if (kt + TILE_K < p.K) {
            ffm_stage_tile(A, smA[nxt], i0, p.M, p.lda, kt + TILE_K, p.K, p.transA, tid);
            ffm_stage_tile(B, smB[nxt], j0, p.N, p.ldb, kt + TILE_K, p.K, p.transB, tid);
        }

        const h16* sa = smA[cur];
        const h16* sb = smB[cur];
        v16h afrag[2], bfrag[4];
        #pragma unroll
        for (int ai = 0; ai < 2; ++ai)
            afrag[ai] = ffm_load_frag(sa, wrow * 32 + ai * 16 + l16, k0sel);
        #pragma unroll
        for (int bj = 0; bj < 4; ++bj)
            bfrag[bj] = ffm_load_frag(sb, wcol * 64 + bj * 16 + l16, k0sel);

        #pragma unroll
        for (int ai = 0; ai < 2; ++ai)
            #pragma unroll
            for (int bj = 0; bj < 4; ++bj)
                acc[ai][bj] = __builtin_amdgcn_wmma_f32_16x16x32_f16(
                    false, afrag[ai], false, bfrag[bj],
                    (short)0, acc[ai][bj], false, false);

        ffm_stage_fence();
        cur = nxt;
    }

    // Epilogue. C layout: element e of v8f -> row = base + e + 8*half, col = l16.
    #pragma unroll
    for (int bj = 0; bj < 4; ++bj) {
        int Cc = j0 + wcol * 64 + bj * 16 + l16;
        if (Cc >= p.N) continue;
        float bv = (p.flags & EPI_BIAS) ? p.bias[Cc] : 0.0f;
        #pragma unroll
        for (int ai = 0; ai < 2; ++ai) {
            #pragma unroll
            for (int e = 0; e < 8; ++e) {
                int R = i0 + wrow * 32 + ai * 16 + e + 8 * half;
                if (R >= p.M) continue;
                float v = acc[ai][bj][e] + bv;
                if (p.flags & EPI_GELU) v = ffm_gelu(v);
                if (p.flags & EPI_EXP)  v = expf(v - rowAux[R]) * p.scale;
                if (p.flags & EPI_DIV)  v = v / (rowAux[R] + 1e-8f);
                if (p.flags & EPI_RES)  v += p.res[(long long)R * p.ldres + Cc];
                if ((p.flags & EPI_OUT32) && Cc >= p.c32off)
                    C32[(long long)R * p.ldc32 + (Cc - p.c32off)] = v;
                if (p.flags & EPI_OUT16)
                    C16[(long long)R * p.ldc16 + Cc] = (h16)v;
            }
        }
    }
}

// ---- helper kernels -------------------------------------------------------

__global__ void ffm_concat_f16(const float* __restrict__ a,
                               const float* __restrict__ b,
                               h16* __restrict__ out, long long rows) {
    long long idx = (long long)blockIdx.x * blockDim.x + threadIdx.x;
    if (idx >= rows * 768) return;
    long long r = idx / 768;
    int c = (int)(idx - r * 768);
    float v = (c < 384) ? a[r * 384 + c] : b[r * 384 + (c - 384)];
    out[idx] = (h16)v;
}

// src (K,N) f32 -> dst (N,K) f16
__global__ void ffm_transpose_f16(const float* __restrict__ src,
                                  h16* __restrict__ dst, int K, int N) {
    long long idx = (long long)blockIdx.x * blockDim.x + threadIdx.x;
    if (idx >= (long long)K * N) return;
    int n = (int)(idx / K);
    int k = (int)(idx - (long long)n * K);
    dst[idx] = (h16)src[(long long)k * N + n];
}

__global__ void ffm_convert_f16(const float* __restrict__ src,
                                h16* __restrict__ dst, long long n) {
    long long idx = (long long)blockIdx.x * blockDim.x + threadIdx.x;
    if (idx < n) dst[idx] = (h16)src[idx];
}

__global__ __launch_bounds__(128)
void ffm_layernorm(const float* __restrict__ x, const float* __restrict__ g,
                   const float* __restrict__ bb, h16* __restrict__ out, int D) {
    long long row = blockIdx.x;
    const float* xr = x + row * D;
    __shared__ float s1[128], s2[128];
    float s = 0.f, ss = 0.f;
    for (int c = threadIdx.x; c < D; c += 128) {
        float v = xr[c]; s += v; ss += v * v;
    }
    s1[threadIdx.x] = s; s2[threadIdx.x] = ss;
    __syncthreads();
    for (int off = 64; off; off >>= 1) {
        if (threadIdx.x < off) {
            s1[threadIdx.x] += s1[threadIdx.x + off];
            s2[threadIdx.x] += s2[threadIdx.x + off];
        }
        __syncthreads();
    }
    float mu  = s1[0] / D;
    float var = s2[0] / D - mu * mu;
    float rs  = rsqrtf(var + 1e-5f);
    for (int c = threadIdx.x; c < D; c += 128)
        out[row * D + c] = (h16)((xr[c] - mu) * rs * g[c] + bb[c]);
}

// out[row] = 0.5 * sum_c src[row*ld + c]^2
__global__ __launch_bounds__(64)
void ffm_row_halfsumsq(const h16* __restrict__ src, int ld, int cols,
                       float* __restrict__ out) {
    long long row = blockIdx.x;
    const h16* r = src + row * (long long)ld;
    __shared__ float sm[64];
    float s = 0.f;
    for (int c = threadIdx.x; c < cols; c += 64) {
        float v = (float)r[c]; s += v * v;
    }
    sm[threadIdx.x] = s;
    __syncthreads();
    for (int off = 32; off; off >>= 1) {
        if (threadIdx.x < off) sm[threadIdx.x] += sm[threadIdx.x + off];
        __syncthreads();
    }
    if (threadIdx.x == 0) out[row] = 0.5f * sm[0];
}

// skp[b, m] = sum_t kp[b*T + t, m]
__global__ void ffm_colsum_batch(const h16* __restrict__ kp,
                                 float* __restrict__ skp, int T, int m) {
    int b = blockIdx.x, mi = threadIdx.x;
    const h16* p = kp + (long long)b * T * m + mi;
    float s = 0.f;
    for (int t = 0; t < T; ++t) s += (float)p[(long long)t * m];
    skp[b * m + mi] = s;
}

// Dn[row] = sum_m qp[row, m] * skp[row/T, m]   (wave per row)
__global__ __launch_bounds__(256)
void ffm_dn(const h16* __restrict__ qp, const float* __restrict__ skp,
            float* __restrict__ Dn, long long Nt, int T, int m) {
    long long row = (long long)blockIdx.x * 8 + (threadIdx.x >> 5);
    int lane = threadIdx.x & 31;
    if (row >= Nt) return;
    int b = (int)(row / T);
    float s = 0.f;
    for (int j = lane; j < m; j += 32)
        s += (float)qp[row * m + j] * skp[b * m + j];
    for (int off = 16; off; off >>= 1)
        s += __shfl_xor(s, off, 32);
    if (lane == 0) Dn[row] = s;
}

// ---- launch ---------------------------------------------------------------

extern "C" void kernel_launch(void* const* d_in, const int* in_sizes, int n_in,
                              void* d_out, int out_size, void* d_ws, size_t ws_size,
                              hipStream_t stream) {
    (void)in_sizes; (void)n_in; (void)out_size; (void)ws_size;
    const int Bsz = 16, T = 3136, Cin = 768, D = 384, Mfeat = 192;
    const long long Nt = (long long)Bsz * T;           // 50176
    const int KQV = 3 * D;                             // 1152

    const float* input_1 = (const float*)d_in[0];
    const float* input_2 = (const float*)d_in[1];
    const float* W1      = (const float*)d_in[2];
    const float* b1      = (const float*)d_in[3];
    const float* W2      = (const float*)d_in[4];
    const float* b2      = (const float*)d_in[5];
    const float* kqv_w   = (const float*)d_in[6];
    const float* kqv_b   = (const float*)d_in[7];
    const float* proj_w  = (const float*)d_in[8];
    const float* proj_b  = (const float*)d_in[9];
    const float* ln1_g   = (const float*)d_in[10];
    const float* ln1_b   = (const float*)d_in[11];
    const float* ln2_g   = (const float*)d_in[12];
    const float* ln2_b   = (const float*)d_in[13];
    const float* mlp_w1  = (const float*)d_in[14];
    const float* mlp_b1  = (const float*)d_in[15];
    const float* mlp_w2  = (const float*)d_in[16];
    const float* mlp_b2  = (const float*)d_in[17];
    const float* w_prm   = (const float*)d_in[18];
    float* out = (float*)d_out;

    // scratch layout (all 256B aligned)
    char* ws = (char*)d_ws;
    size_t off = 0;
    auto take = [&](size_t bytes) { size_t o = off; off += (bytes + 255) & ~(size_t)255; return o; };
    const size_t oW1T  = take((size_t)D * Cin * 2);       // (384,768)
    const size_t oW2T  = take((size_t)D * D * 2);
    const size_t oKQVT = take((size_t)KQV * D * 2);       // (1152,384)
    const size_t oPRJT = take((size_t)D * D * 2);
    const size_t oM1T  = take((size_t)D * D * 2);
    const size_t oM2T  = take((size_t)D * D * 2);
    const size_t oWPRM = take((size_t)Mfeat * D * 2);     // (192,384)
    const size_t oKXD  = take((size_t)Nt * 4);
    const size_t oQXD  = take((size_t)Nt * 4);
    const size_t oDN   = take((size_t)Nt * 4);
    const size_t oSKP  = take((size_t)Bsz * Mfeat * 4);
    const size_t oKPTV = take((size_t)Bsz * D * Mfeat * 2);
    const size_t oR1   = take((size_t)Nt * Cin * 2);      // xin(f16) -> x(f32) -> v(f32)
    const size_t oRA   = take((size_t)Nt * D * 2);        // h1 -> ya -> z
    const size_t oRB   = take((size_t)Nt * D * 2);        // xn -> mlp_h
    const size_t oKQV  = take((size_t)Nt * KQV * 2);      // kqv f16, later y(f32) reuse
    const size_t oKP   = take((size_t)Nt * Mfeat * 2);
    const size_t oQP   = take((size_t)Nt * Mfeat * 2);

    h16*   wts_w1t  = (h16*)(ws + oW1T);
    h16*   wts_w2t  = (h16*)(ws + oW2T);
    h16*   wts_kqvt = (h16*)(ws + oKQVT);
    h16*   wts_prjt = (h16*)(ws + oPRJT);
    h16*   wts_m1t  = (h16*)(ws + oM1T);
    h16*   wts_m2t  = (h16*)(ws + oM2T);
    h16*   wts_prm  = (h16*)(ws + oWPRM);
    float* kxd  = (float*)(ws + oKXD);
    float* qxd  = (float*)(ws + oQXD);
    float* Dn   = (float*)(ws + oDN);
    float* skp  = (float*)(ws + oSKP);
    h16*   kptv = (h16*)(ws + oKPTV);
    h16*   xin  = (h16*)(ws + oR1);
    float* x32  = (float*)(ws + oR1);   // reuse after xin dead
    float* v32  = (float*)(ws + oR1);   // reuse after x32 dead
    h16*   h1   = (h16*)(ws + oRA);
    h16*   ya   = (h16*)(ws + oRA);
    h16*   z16  = (h16*)(ws + oRA);
    h16*   xn   = (h16*)(ws + oRB);
    h16*   hml  = (h16*)(ws + oRB);
    h16*   kqv16 = (h16*)(ws + oKQV);
    float* y32   = (float*)(ws + oKQV); // reuse after kqv16 dead
    h16*   kp16 = (h16*)(ws + oKP);
    h16*   qp16 = (h16*)(ws + oQP);

    auto cdiv = [](long long a, long long b) { return (unsigned)((a + b - 1) / b); };

    // S0: weight transposes + converts
    ffm_transpose_f16<<<cdiv((long long)Cin * D, 256), 256, 0, stream>>>(W1, wts_w1t, Cin, D);
    ffm_transpose_f16<<<cdiv((long long)D * D, 256), 256, 0, stream>>>(W2, wts_w2t, D, D);
    ffm_transpose_f16<<<cdiv((long long)D * KQV, 256), 256, 0, stream>>>(kqv_w, wts_kqvt, D, KQV);
    ffm_transpose_f16<<<cdiv((long long)D * D, 256), 256, 0, stream>>>(proj_w, wts_prjt, D, D);
    ffm_transpose_f16<<<cdiv((long long)D * D, 256), 256, 0, stream>>>(mlp_w1, wts_m1t, D, D);
    ffm_transpose_f16<<<cdiv((long long)D * D, 256), 256, 0, stream>>>(mlp_w2, wts_m2t, D, D);
    ffm_convert_f16<<<cdiv((long long)Mfeat * D, 256), 256, 0, stream>>>(w_prm, wts_prm, (long long)Mfeat * D);

    // S1: concat -> f16
    ffm_concat_f16<<<cdiv(Nt * Cin, 256), 256, 0, stream>>>(input_1, input_2, xin, Nt);

    auto gemm = [&](const h16* A, const h16* Bm, float* C32, h16* C16,
                    const float* bias, const float* rowAux, const float* res,
                    int M, int N, int K, int lda, int ldb, int ldc32, int ldc16,
                    int ldres, long long sA, long long sB, long long sC32,
                    long long sC16, long long sAux, int tA, int tB,
                    int flags, float scale, int c32off, int batches) {
        GemmParams p{A, Bm, C32, C16, bias, rowAux, res, M, N, K,
                     lda, ldb, ldc32, ldc16, ldres, sA, sB, sC32, sC16, sAux,
                     tA, tB, flags, c32off, scale};
        dim3 g(cdiv(N, TILE_N), cdiv(M, TILE_M), batches);
        ffm_gemm_wmma<<<g, 256, 0, stream>>>(p);
    };

    // S2: h1 = gelu(xin @ W1 + b1)  (f16)
    gemm(xin, wts_w1t, nullptr, h1, b1, nullptr, nullptr,
         (int)Nt, D, Cin, Cin, Cin, 0, D, 0, 0, 0, 0, 0, 0, 0, 0,
         EPI_BIAS | EPI_GELU | EPI_OUT16, 1.f, 0, 1);

    // S3: x = h1 @ W2 + b2  (f32)
    gemm(h1, wts_w2t, x32, nullptr, b2, nullptr, nullptr,
         (int)Nt, D, D, D, D, D, 0, 0, 0, 0, 0, 0, 0, 0, 0,
         EPI_BIAS | EPI_OUT32, 1.f, 0, 1);

    // S4: xn = LN1(x)  (f16)
    ffm_layernorm<<<(unsigned)Nt, 128, 0, stream>>>(x32, ln1_g, ln1_b, xn, D);

    // S5: kqv = xn @ kqv_w + b  (f16 full; f32 for v slice cols 768..1151)
    gemm(xn, wts_kqvt, v32, kqv16, kqv_b, nullptr, nullptr,
         (int)Nt, KQV, D, D, D, D, KQV, 0, 0, 0, 0, 0, 0, 0, 0,
         EPI_BIAS | EPI_OUT16 | EPI_OUT32, 1.f, 2 * D, 1);

    // S6: kxd/qxd = 0.5*|k|^2, 0.5*|q|^2 per row
    ffm_row_halfsumsq<<<(unsigned)Nt, 64, 0, stream>>>(kqv16, KQV, D, kxd);
    ffm_row_halfsumsq<<<(unsigned)Nt, 64, 0, stream>>>(kqv16 + D, KQV, D, qxd);

    const float inv_sqrt_m = 0.072168783648703221f; // 1/sqrt(192)
    // S7: kp = exp(k @ w_prm^T - kxd) / sqrt(m)
    gemm(kqv16, wts_prm, nullptr, kp16, nullptr, kxd, nullptr,
         (int)Nt, Mfeat, D, KQV, D, 0, Mfeat, 0, 0, 0, 0, 0, 0, 0, 0,
         EPI_EXP | EPI_OUT16, inv_sqrt_m, 0, 1);
    // S8: qp likewise
    gemm(kqv16 + D, wts_prm, nullptr, qp16, nullptr, qxd, nullptr,
         (int)Nt, Mfeat, D, KQV, D, 0, Mfeat, 0, 0, 0, 0, 0, 0, 0, 0,
         EPI_EXP | EPI_OUT16, inv_sqrt_m, 0, 1);

    // S9: skp[b,m] = sum_t kp ;  Dn[row] = qp . skp
    ffm_colsum_batch<<<Bsz, Mfeat, 0, stream>>>(kp16, skp, T, Mfeat);
    ffm_dn<<<cdiv(Nt, 8), 256, 0, stream>>>(qp16, skp, Dn, Nt, T, Mfeat);

    // S10: kptv[b] = v^T @ kp   (M=D, N=m, K=T; both operands transposed)
    gemm(kqv16 + 2 * D, kp16, nullptr, kptv, nullptr, nullptr, nullptr,
         D, Mfeat, T, KQV, Mfeat, 0, Mfeat, 0,
         (long long)T * KQV, (long long)T * Mfeat, 0, (long long)D * Mfeat, 0,
         1, 1, EPI_OUT16, 1.f, 0, Bsz);

    // S11: ya[b] = (qp @ kptv^T) / (Dn + 1e-8)   (M=T, N=D, K=m)
    gemm(qp16, kptv, nullptr, ya, nullptr, Dn, nullptr,
         T, D, Mfeat, Mfeat, Mfeat, 0, D, 0,
         (long long)T * Mfeat, (long long)D * Mfeat, 0, (long long)T * D, T,
         0, 0, EPI_DIV | EPI_OUT16, 1.f, 0, Bsz);

    // S12: y = v + ya @ proj_w + proj_b   (f32)
    gemm(ya, wts_prjt, y32, nullptr, proj_b, nullptr, v32,
         (int)Nt, D, D, D, D, D, 0, D, 0, 0, 0, 0, 0, 0, 0,
         EPI_BIAS | EPI_RES | EPI_OUT32, 1.f, 0, 1);

    // S13: z = LN2(y)  (f16)
    ffm_layernorm<<<(unsigned)Nt, 128, 0, stream>>>(y32, ln2_g, ln2_b, z16, D);

    // S14: hml = gelu(z @ mlp_w1 + b)  (f16)
    gemm(z16, wts_m1t, nullptr, hml, mlp_b1, nullptr, nullptr,
         (int)Nt, D, D, D, D, 0, D, 0, 0, 0, 0, 0, 0, 0, 0,
         EPI_BIAS | EPI_GELU | EPI_OUT16, 1.f, 0, 1);

    // S15: out = y + hml @ mlp_w2 + b  (f32 -> d_out)
    gemm(hml, wts_m2t, out, nullptr, mlp_b2, nullptr, y32,
         (int)Nt, D, D, D, D, D, 0, D, 0, 0, 0, 0, 0, 0, 0,
         EPI_BIAS | EPI_RES | EPI_OUT32, 1.f, 0, 1);
}
